// SchemaAugmentedSOMT_10685878632503
// MI455X (gfx1250) — matrix operand order repaired and checked by reading, not compiled, not run on hardware
//
#include <hip/hip_runtime.h>
#include <math.h>

// ---------------- CDNA5 WMMA types ----------------
typedef __attribute__((ext_vector_type(16))) __bf16 v16bf;
typedef __attribute__((ext_vector_type(2)))  __bf16 v2bf;
typedef __attribute__((ext_vector_type(8)))  float  v8f;

union Frag {
  v16bf v;
  uint4 q[2];
};

// native f32 -> bf16 (RNE); lowers to v_cvt_pk_bf16_f32 / v_cvt_bf16
__device__ __forceinline__ unsigned short f2bf(float f) {
  union { __bf16 h; unsigned short u; } r;
  r.h = (__bf16)f;
  return r.u;
}
__device__ __forceinline__ unsigned int pk2bf(float a, float b) {
  v2bf h = { (__bf16)a, (__bf16)b };
  return __builtin_bit_cast(unsigned int, h);
}

// ---------------- problem constants ----------------
#define V_    50257
#define D_    512
#define H_    8
#define NL_   4
#define S_    64
#define B_    2
#define L_    512
#define M_    1024
#define NEGF  (-1e30f)

static const int GBM = 128;  // gemm tile M  (all calls: M % 128 == 0)
static const int GBN = 128;  // gemm tile N  (N may be ragged)
static const int GBK = 32;   // gemm tile K  (all calls: K % 32 == 0)
#define LDA_S 40             // LDS A row stride (ushorts)
#define LDB_S 136            // LDS B row stride (ushorts)

// ======================================================================
// Generic bf16 WMMA GEMM:  C[z] = act( alpha * A[z] * op(W[z]) + bias [+ C] )
//   A : [M,K] fp32 row-major, row stride lda, z offset = zo*sAo + zi*sAi
//   W : if transB: B[k][n] = W[n*ldw + k]   (the w.T of _lin)
//       else     : B[k][n] = W[k*ldw + n]
//   zo = z / zdiv ; zi = z % zdiv
// Requires K % 32 == 0, M % 128 == 0, and (non-transB) N >= 4.
// N edges handled via clamped addresses + selects (no exec-mask loads).
// 8 waves in 4(M) x 2(N); each wave computes 32x64 = 2x4 wmma tiles.
// ======================================================================
__global__ __launch_bounds__(256) void gemm_bf16_kernel(
    const float* __restrict__ A, long lda, long sAo, long sAi,
    const float* __restrict__ W, long ldw, long sWo, long sWi,
    const float* __restrict__ bias,
    float* __restrict__ C, long ldc, long sCo, long sCi,
    int M, int N, int K, int zdiv, int transB, int act, int accum, float alpha)
{
  __shared__ unsigned short sA[128 * LDA_S];
  __shared__ unsigned short sB[32 * LDB_S];

  int z  = blockIdx.z;
  int zo = z / zdiv, zi = z % zdiv;
  A += zo * sAo + zi * sAi;
  W += zo * sWo + zi * sWi;
  C += zo * sCo + zi * sCi;

  int m0 = blockIdx.y * GBM;
  int n0 = blockIdx.x * GBN;
  int tid  = threadIdx.x;
  int wave = tid >> 5;
  int lane = tid & 31;
  int wm = wave >> 1;   // 0..3  (M dir, 32 rows each)
  int wn = wave & 1;    // 0..1  (N dir, 64 cols each)

  v8f acc[2][4];
#pragma unroll
  for (int i = 0; i < 2; ++i)
#pragma unroll
    for (int j = 0; j < 4; ++j) acc[i][j] = (v8f)(0.f);

  for (int k0 = 0; k0 < K; k0 += GBK) {
    float4 fa[4];
    // ---- phase 1: issue all A loads (128x32 tile, 1024 float4 chunks) ----
#pragma unroll
    for (int i = 0; i < 4; ++i) {
      int idx = tid + i * 256;
      int m = idx >> 3, k4 = (idx & 7) * 4;
      int gm = m0 + m;
      if (gm >= M) gm = M - 1;
      fa[i] = *(const float4*)(A + (long)gm * lda + k0 + k4);
    }
    // ---- phase 2: B loads + A pack/store ----
    if (transB) {
      float4 fb[4];
      bool   okb[4];
#pragma unroll
      for (int i = 0; i < 4; ++i) {
        int idx = tid + i * 256;            // chunks along k
        int n = idx >> 3, k4 = (idx & 7) * 4;
        int gn = n0 + n;
        okb[i] = (gn < N);
        if (!okb[i]) gn = N - 1;
        fb[i] = *(const float4*)(W + (long)gn * ldw + k0 + k4);
      }
#pragma unroll
      for (int i = 0; i < 4; ++i) {
        int idx = tid + i * 256;
        int m = idx >> 3, k4 = (idx & 7) * 4;
        uint2 p; p.x = pk2bf(fa[i].x, fa[i].y); p.y = pk2bf(fa[i].z, fa[i].w);
        *(uint2*)&sA[m * LDA_S + k4] = p;
      }
#pragma unroll
      for (int i = 0; i < 4; ++i) {
        int idx = tid + i * 256;
        int n = idx >> 3, k4 = (idx & 7) * 4;
        bool ok = okb[i];
        sB[(k4 + 0) * LDB_S + n] = ok ? f2bf(fb[i].x) : (unsigned short)0;
        sB[(k4 + 1) * LDB_S + n] = ok ? f2bf(fb[i].y) : (unsigned short)0;
        sB[(k4 + 2) * LDB_S + n] = ok ? f2bf(fb[i].z) : (unsigned short)0;
        sB[(k4 + 3) * LDB_S + n] = ok ? f2bf(fb[i].w) : (unsigned short)0;
      }
    } else {
      float4 fb[4];
      int    shb[4];
#pragma unroll
      for (int i = 0; i < 4; ++i) {
        int idx = tid + i * 256;            // chunks along n
        int k = idx >> 5, n4 = (idx & 31) * 4;
        int gn = n0 + n4;
        int cb = (gn > N - 4) ? (N - 4) : gn;   // clamp base (N >= 4)
        shb[i] = gn - cb;
        fb[i] = *(const float4*)(W + (long)(k0 + k) * ldw + cb);
      }
#pragma unroll
      for (int i = 0; i < 4; ++i) {
        int idx = tid + i * 256;
        int m = idx >> 3, k4 = (idx & 7) * 4;
        uint2 p; p.x = pk2bf(fa[i].x, fa[i].y); p.y = pk2bf(fa[i].z, fa[i].w);
        *(uint2*)&sA[m * LDA_S + k4] = p;
      }
#pragma unroll
      for (int i = 0; i < 4; ++i) {
        int idx = tid + i * 256;
        int k = idx >> 5, n4 = (idx & 31) * 4;
        if (shb[i] == 0) {
          uint2 p; p.x = pk2bf(fb[i].x, fb[i].y); p.y = pk2bf(fb[i].z, fb[i].w);
          *(uint2*)&sB[k * LDB_S + n4] = p;
        } else {
          // straddling chunk at the N edge: shift + zero-fill
          float vv[4] = { fb[i].x, fb[i].y, fb[i].z, fb[i].w };
#pragma unroll
          for (int e = 0; e < 4; ++e) {
            int src = e + shb[i];
            float v = (src < 4) ? vv[src] : 0.f;
            sB[k * LDB_S + n4 + e] = f2bf(v);
          }
        }
      }
    }
    __syncthreads();

    // ---- fragments per ISA layout ----
    int g  = lane >> 4;     // lane group
    int am = lane & 15;     // A row within subtile
    Frag a[2], b[4];
#pragma unroll
    for (int i = 0; i < 2; ++i) {
      const unsigned short* row = sA + (wm * 32 + i * 16 + am) * LDA_S;
      a[i].q[0] = *(const uint4*)(row + g * 8);
      a[i].q[1] = *(const uint4*)(row + 16 + g * 8);
    }
    int bn = lane & 15;  // B column (N index)
#pragma unroll
    for (int j = 0; j < 4; ++j) {
      const unsigned short* col = sB + wn * 64 + j * 16 + bn;
#pragma unroll
      for (int s = 0; s < 16; ++s) {
        int kk = (s < 8) ? (g * 8 + s) : (16 + g * 8 + (s - 8));
        b[j].v[s] = __builtin_bit_cast(__bf16, col[kk * LDB_S]);
      }
    }
#pragma unroll
    for (int i = 0; i < 2; ++i)
#pragma unroll
      for (int j = 0; j < 4; ++j)
        acc[i][j] = __builtin_amdgcn_wmma_f32_16x16x32_bf16(
            false, a[i].v, false, b[j].v, (short)0, acc[i][j], false, false);
    __syncthreads();
  }

  // ---- epilogue: C/D layout lane = (row>=8)?16:0 + col ; vgpr r = row&7 ----
  int cn = lane & 15, cg = lane >> 4;
#pragma unroll
  for (int i = 0; i < 2; ++i)
#pragma unroll
    for (int j = 0; j < 4; ++j) {
      int col = n0 + wn * 64 + j * 16 + cn;
      if (col >= N) continue;
      float bv = bias ? bias[col] : 0.f;
#pragma unroll
      for (int r = 0; r < 8; ++r) {
        int row = m0 + wm * 32 + i * 16 + cg * 8 + r;
        if (row >= M) continue;
        float v = acc[i][j][r] * alpha + bv;
        long off = (long)row * ldc + col;
        if (accum) v += C[off];
        if (act == 1) v = fmaxf(v, 0.f);
        C[off] = v;
      }
    }
}

// ======================================================================
// helper kernels
// ======================================================================
__global__ void embed_kernel(const int* __restrict__ x, const float* __restrict__ embW,
                             const float* __restrict__ pos, float* __restrict__ xemb,
                             float* __restrict__ h) {
  int tk = blockIdx.x;
  int l = tk & (L_ - 1);
  int tok = x[tk];
  for (int d = threadIdx.x; d < D_; d += blockDim.x) {
    float v = embW[(long)tok * D_ + d] + pos[l * D_ + d];
    xemb[(long)tk * D_ + d] = v;
    h[(long)tk * D_ + d]    = v;
  }
}

__global__ __launch_bounds__(256) void softmax_causal_kernel(float* __restrict__ sc) {
  __shared__ float red[256];
  long base = (long)blockIdx.x * L_;
  int i = blockIdx.x & (L_ - 1);
  int t = threadIdx.x;
  float v0 = sc[base + t];
  float v1 = sc[base + t + 256];
  if (t > i)       v0 = NEGF;
  if (t + 256 > i) v1 = NEGF;
  red[t] = fmaxf(v0, v1); __syncthreads();
  for (int s = 128; s > 0; s >>= 1) { if (t < s) red[t] = fmaxf(red[t], red[t + s]); __syncthreads(); }
  float m = red[0]; __syncthreads();
  float e0 = __expf(v0 - m), e1 = __expf(v1 - m);
  red[t] = e0 + e1; __syncthreads();
  for (int s = 128; s > 0; s >>= 1) { if (t < s) red[t] += red[t + s]; __syncthreads(); }
  float inv = 1.f / red[0];
  sc[base + t]       = e0 * inv;
  sc[base + t + 256] = e1 * inv;
}

__global__ __launch_bounds__(256) void ln_residual_kernel(
    const float* __restrict__ x, const float* __restrict__ y,
    const float* __restrict__ g, const float* __restrict__ bt, float* __restrict__ out) {
  __shared__ float red[256];
  long base = (long)blockIdx.x * D_;
  int t = threadIdx.x;
  float a0 = x[base + t] + y[base + t];
  float a1 = x[base + t + 256] + y[base + t + 256];
  red[t] = a0 + a1; __syncthreads();
  for (int s = 128; s > 0; s >>= 1) { if (t < s) red[t] += red[t + s]; __syncthreads(); }
  float mean = red[0] * (1.f / D_); __syncthreads();
  float d0 = a0 - mean, d1 = a1 - mean;
  red[t] = d0 * d0 + d1 * d1; __syncthreads();
  for (int s = 128; s > 0; s >>= 1) { if (t < s) red[t] += red[t + s]; __syncthreads(); }
  float inv = rsqrtf(red[0] * (1.f / D_) + 1e-5f);
  out[base + t]       = d0 * inv * g[t] + bt[t];
  out[base + t + 256] = d1 * inv * g[t + 256] + bt[t + 256];
}

// generic tiny 2-layer MLP: out = finalAct( relu(in*w1^T+b1) * w2^T + b2 )
__global__ __launch_bounds__(128) void small_mlp_kernel(
    const float* __restrict__ in, long inStride,
    const float* __restrict__ extraA, const float* __restrict__ extraB,
    const float* __restrict__ w1, const float* __restrict__ b1, int K, int Hd,
    const float* __restrict__ w2, const float* __restrict__ b2, int Nout,
    float* __restrict__ out, long outStride, int finalAct, int ntok)
{
  int t = blockIdx.x;
  if (t >= ntok) return;
  __shared__ float sin[520];
  __shared__ float sh[128];
  for (int d = threadIdx.x; d < K; d += 128) sin[d] = in[(long)t * inStride + d];
  int Ktot = K;
  if (extraA) {
    if (threadIdx.x == 0) { sin[K] = extraA[t]; sin[K + 1] = extraB[t]; }
    Ktot = K + 2;
  }
  __syncthreads();
  for (int j = threadIdx.x; j < Hd; j += 128) {
    float s = b1[j];
    const float* wr = w1 + (long)j * Ktot;
    for (int d = 0; d < Ktot; ++d) s += wr[d] * sin[d];
    sh[j] = fmaxf(s, 0.f);
  }
  __syncthreads();
  for (int o = threadIdx.x; o < Nout; o += 128) {
    float s = b2[o];
    const float* wr = w2 + (long)o * Hd;
    for (int d = 0; d < Hd; ++d) s += wr[d] * sh[d];
    if (finalAct == 1) s = 1.f / (1.f + __expf(-s));
    out[(long)t * outStride + o] = s;
  }
}

// online (max, Z, S=sum e^(x-m)(x-m)) -> entropy = log Z - S/Z
__global__ __launch_bounds__(256) void entropy_kernel(const float* __restrict__ logits,
                                                      float* __restrict__ ent) {
  __shared__ float sm[256], sz[256], ss[256];
  int tk = blockIdx.x, t = threadIdx.x;
  const float* row = logits + (long)tk * V_;
  float m = NEGF, z = 0.f, s = 0.f;
  for (int j = t; j < V_; j += 256) {
    float x = row[j];
    if (x <= m) { float e = __expf(x - m); z += e; s += (x - m) * e; }
    else { float c = __expf(m - x); s = c * (s + (m - x) * z); z = c * z + 1.f; m = x; }
  }
  sm[t] = m; sz[t] = z; ss[t] = s; __syncthreads();
  for (int k = 128; k > 0; k >>= 1) {
    if (t < k) {
      float m2 = sm[t + k], z2 = sz[t + k], s2 = ss[t + k];
      float Mx = fmaxf(sm[t], m2);
      float c1 = __expf(sm[t] - Mx), c2 = __expf(m2 - Mx);
      ss[t] = c1 * (ss[t] + (sm[t] - Mx) * sz[t]) + c2 * (s2 + (m2 - Mx) * z2);
      sz[t] = c1 * sz[t] + c2 * z2;
      sm[t] = Mx;
    }
    __syncthreads();
  }
  if (t == 0) ent[tk] = logf(sz[0]) - ss[0] / sz[0];
}

__global__ __launch_bounds__(256) void entnorm_kernel(const float* __restrict__ ent,
                                                      float* __restrict__ enorm) {
  __shared__ float red[256];
  int b = blockIdx.x, t = threadIdx.x;
  float e0 = ent[b * L_ + t], e1 = ent[b * L_ + t + 256];
  red[t] = e0 + e1; __syncthreads();
  for (int s = 128; s > 0; s >>= 1) { if (t < s) red[t] += red[t + s]; __syncthreads(); }
  float mean = red[0] * (1.f / L_); __syncthreads();
  float d0 = e0 - mean, d1 = e1 - mean;
  red[t] = d0 * d0 + d1 * d1; __syncthreads();
  for (int s = 128; s > 0; s >>= 1) { if (t < s) red[t] += red[t + s]; __syncthreads(); }
  float stdv = sqrtf(red[0] * (1.f / L_)) + 1e-12f;
  enorm[b * L_ + t]       = 1.f / (1.f + __expf(-d0 / stdv));
  enorm[b * L_ + t + 256] = 1.f / (1.f + __expf(-d1 / stdv));
}

__global__ void scan_kernel(const float* __restrict__ en, float* __restrict__ cmean,
                            float* __restrict__ cmax) {
  int b = blockIdx.x;
  if (threadIdx.x == 0) {
    float s = 0.f, mx = NEGF;
    for (int i = 0; i < L_; ++i) {
      float v = en[b * L_ + i];
      s += v; mx = fmaxf(mx, v);
      cmean[b * L_ + i] = s / (float)(i + 1);
      cmax[b * L_ + i]  = mx;
    }
  }
}

__global__ void zero2_kernel(float* p) { if (threadIdx.x < 2) p[threadIdx.x] = 0.f; }

__global__ __launch_bounds__(256) void mask_stats_kernel(
    const float* __restrict__ enorm, const float* __restrict__ dthr,
    const float* __restrict__ ck, const float* __restrict__ cv,
    float* __restrict__ wmask, float* __restrict__ outscal) {
  __shared__ float red[256];
  int tk = blockIdx.x, t = threadIdx.x;
  long base = (long)tk * D_;
  float k0 = ck[base + t], k1 = ck[base + t + 256];
  float v0 = cv[base + t], v1 = cv[base + t + 256];
  red[t] = k0 * k0 + k1 * k1; __syncthreads();
  for (int s = 128; s > 0; s >>= 1) { if (t < s) red[t] += red[t + s]; __syncthreads(); }
  float nk = red[0]; __syncthreads();
  red[t] = v0 * v0 + v1 * v1; __syncthreads();
  for (int s = 128; s > 0; s >>= 1) { if (t < s) red[t] += red[t + s]; __syncthreads(); }
  float nv = red[0];
  if (t == 0) {
    float en = enorm[tk];
    float msk = en > dthr[tk] ? 1.f : 0.f;
    wmask[tk] = msk;
    if (msk > 0.f) {
      atomicAdd(outscal + 0, en);
      atomicAdd(outscal + 1, 0.5f * (sqrtf(nk) + sqrtf(nv)));
    }
  }
}

__global__ __launch_bounds__(256) void retr_softmax_kernel(
    float* __restrict__ sc, const float* __restrict__ age, const float* __restrict__ wmask) {
  __shared__ float red[256];
  int b = blockIdx.x >> 9;
  int q = blockIdx.x & (L_ - 1);
  float* row = sc + (long)blockIdx.x * (M_ + L_);
  int t = threadIdx.x;
  float v[6];
  float m = NEGF;
#pragma unroll
  for (int jj = 0; jj < 6; ++jj) {
    int j = t + jj * 256;
    float x = row[j];
    if (j < M_) x -= 0.01f * age[b * M_ + j];
    else {
      int l = j - M_;
      if (!(l <= q && wmask[b * L_ + l] > 0.5f)) x = NEGF;
    }
    v[jj] = x; m = fmaxf(m, x);
  }
  red[t] = m; __syncthreads();
  for (int s = 128; s > 0; s >>= 1) { if (t < s) red[t] = fmaxf(red[t], red[t + s]); __syncthreads(); }
  m = red[0]; __syncthreads();
  float zs = 0.f;
#pragma unroll
  for (int jj = 0; jj < 6; ++jj) { v[jj] = __expf(v[jj] - m); zs += v[jj]; }
  red[t] = zs; __syncthreads();
  for (int s = 128; s > 0; s >>= 1) { if (t < s) red[t] += red[t + s]; __syncthreads(); }
  float inv = 1.f / red[0];
#pragma unroll
  for (int jj = 0; jj < 6; ++jj) row[t + jj * 256] = v[jj] * inv;
}

__global__ __launch_bounds__(64) void schema_softmax_kernel(
    const float* __restrict__ router, const float* __restrict__ sim, float* __restrict__ attn) {
  __shared__ float red[64];
  int tk = blockIdx.x, s = threadIdx.x;
  float v = router[tk * S_ + s] + sim[tk * S_ + s];
  red[s] = v; __syncthreads();
  for (int k = 32; k > 0; k >>= 1) { if (s < k) red[s] = fmaxf(red[s], red[s + k]); __syncthreads(); }
  float m = red[0]; __syncthreads();
  float e = __expf(v - m);
  red[s] = e; __syncthreads();
  for (int k = 32; k > 0; k >>= 1) { if (s < k) red[s] += red[s + k]; __syncthreads(); }
  attn[tk * S_ + s] = e / red[0];
}

__global__ __launch_bounds__(256) void schema_stats_kernel(
    const float* __restrict__ attn, const float* __restrict__ util,
    const float* __restrict__ us0, const float* __restrict__ uc0, float* __restrict__ outp) {
  __shared__ float ra[256], ru[256];
  int s = blockIdx.x, t = threadIdx.x;
  float sa = 0.f, su = 0.f;
  for (int tk = t; tk < B_ * L_; tk += 256) {
    float a = attn[tk * S_ + s];
    sa += a; su += a * util[tk];
  }
  ra[t] = sa; ru[t] = su; __syncthreads();
  for (int k = 128; k > 0; k >>= 1) { if (t < k) { ra[t] += ra[t + k]; ru[t] += ru[t + k]; } __syncthreads(); }
  if (t == 0) {
    outp[2 + s]      = us0[s] + ru[0];
    outp[2 + S_ + s] = uc0[s] + ra[0];
  }
}

__global__ void qscale_kernel(float* __restrict__ q, const float* __restrict__ unc, int n) {
  int i = blockIdx.x * 256 + threadIdx.x;
  if (i < n) q[i] *= (1.f + unc[i]);
}

__global__ void encmean_kernel(const float* __restrict__ enc, float* __restrict__ out) {
  int b = blockIdx.x;
  for (int d = threadIdx.x; d < D_; d += blockDim.x) {
    float s = 0.f;
    for (int l = 0; l < L_; ++l) s += enc[((long)b * L_ + l) * D_ + d];
    out[b * D_ + d] = s * (1.f / L_);
  }
}

__global__ void delta_kernel(float* __restrict__ fuse, const float* __restrict__ sch,
                             const float* __restrict__ budget, int n) {
  int i = blockIdx.x * 256 + threadIdx.x;
  if (i < n) {
    int b = i >> 18;  // / (L_*D_)
    fuse[i] = fuse[i] * budget[b] + sch[i];
  }
}

__global__ void keep_kernel(const float* __restrict__ imp, const float* __restrict__ obs,
                            const float* __restrict__ age, float* __restrict__ outp, int n) {
  int i = blockIdx.x * 256 + threadIdx.x;
  if (i < n) outp[i] = imp[i] - obs[i] - 0.01f * age[i];
}

// ======================================================================
// host side
// ======================================================================
static inline void launch_gemm(hipStream_t st,
    const float* A, long lda, long sAo, long sAi,
    const float* W, long ldw, long sWo, long sWi,
    const float* bias, float* C, long ldc, long sCo, long sCi,
    int M, int N, int K, int Z, int zdiv, int transB, int act, int accum, float alpha)
{
  dim3 grid((N + GBN - 1) / GBN, (M + GBM - 1) / GBM, Z);
  gemm_bf16_kernel<<<grid, dim3(256), 0, st>>>(A, lda, sAo, sAi, W, ldw, sWo, sWi,
                                               bias, C, ldc, sCo, sCi,
                                               M, N, K, zdiv, transB, act, accum, alpha);
}

extern "C" void kernel_launch(void* const* d_in, const int* in_sizes, int n_in,
                              void* d_out, int out_size, void* d_ws, size_t ws_size,
                              hipStream_t stream) {
  (void)in_sizes; (void)n_in; (void)out_size; (void)ws_size;
  const int*   x      = (const int*)  d_in[0];
  const float* memk   = (const float*)d_in[1];
  const float* memv   = (const float*)d_in[2];
  const float* mage   = (const float*)d_in[3];
  const float* embW   = (const float*)d_in[4];
  const float* pos    = (const float*)d_in[5];
  const float* qkvw   = (const float*)d_in[6];
  const float* qkvb   = (const float*)d_in[7];
  const float* outw   = (const float*)d_in[8];
  const float* outb   = (const float*)d_in[9];
  const float* ff1w   = (const float*)d_in[10];
  const float* ff1b   = (const float*)d_in[11];
  const float* ff2w   = (const float*)d_in[12];
  const float* ff2b   = (const float*)d_in[13];
  const float* ln1w   = (const float*)d_in[14];
  const float* ln1b   = (const float*)d_in[15];
  const float* ln2w   = (const float*)d_in[16];
  const float* ln2b   = (const float*)d_in[17];
  const float* qpw    = (const float*)d_in[18];
  const float* qpb    = (const float*)d_in[19];
  const float* kpw    = (const float*)d_in[20];
  const float* kpb    = (const float*)d_in[21];
  const float* vpw    = (const float*)d_in[22];
  const float* vpb    = (const float*)d_in[23];
  const float* fw     = (const float*)d_in[24];
  const float* fb     = (const float*)d_in[25];
  const float* ug1w   = (const float*)d_in[26];
  const float* ug1b   = (const float*)d_in[27];
  const float* ug2w   = (const float*)d_in[28];
  const float* ug2b   = (const float*)d_in[29];
  const float* thr1w  = (const float*)d_in[30];
  const float* thr1b  = (const float*)d_in[31];
  const float* thr2w  = (const float*)d_in[32];
  const float* thr2b  = (const float*)d_in[33];
  const float* bud1w  = (const float*)d_in[34];
  const float* bud1b  = (const float*)d_in[35];
  const float* bud2w  = (const float*)d_in[36];
  const float* bud2b  = (const float*)d_in[37];
  const float* imp1w  = (const float*)d_in[38];
  const float* imp1b  = (const float*)d_in[39];
  const float* imp2w  = (const float*)d_in[40];
  const float* imp2b  = (const float*)d_in[41];
  const float* obs1w  = (const float*)d_in[42];
  const float* obs1b  = (const float*)d_in[43];
  const float* obs2w  = (const float*)d_in[44];
  const float* obs2b  = (const float*)d_in[45];
  const float* rt1w   = (const float*)d_in[46];
  const float* rt1b   = (const float*)d_in[47];
  const float* rt2w   = (const float*)d_in[48];
  const float* rt2b   = (const float*)d_in[49];
  const float* su1w   = (const float*)d_in[50];
  const float* su1b   = (const float*)d_in[51];
  const float* su2w   = (const float*)d_in[52];
  const float* su2b   = (const float*)d_in[53];
  const float* skeys  = (const float*)d_in[54];
  const float* svals  = (const float*)d_in[55];
  const float* usum0  = (const float*)d_in[56];
  const float* ucnt0  = (const float*)d_in[57];

  float* out = (float*)d_out;
  const long LOGN = (long)B_ * L_ * V_;      // 51,463,168
  float* outTail = out + LOGN;               // [reg, l2, util(64), usage(64), keep(2048)]

  // workspace layout
  float* w = (float*)d_ws;
  const long BLD = (long)B_ * L_ * D_;       // 524288
  long o = 0;
  float* xemb   = w + o; o += BLD;
  float* h      = w + o; o += BLD;
  float* qkv    = w + o; o += (long)B_ * L_ * 3 * D_;       // 1,572,864
  float* scores = w + o; o += (long)B_ * H_ * L_ * L_;      // 4,194,304 (also retrieval scores)
  float* attno  = w + o; o += BLD;
  float* ffh    = w + o; o += (long)B_ * L_ * 4 * D_;       // 2,097,152
  float* tmp    = w + o; o += BLD;
  float* unc    = w + o; o += BLD;
  float* quer   = w + o; o += BLD;
  float* candk  = w + o; o += BLD;
  float* candv  = w + o; o += BLD;
  float* retr   = w + o; o += BLD;
  float* fuseo  = w + o; o += BLD;
  float* scho   = w + o; o += BLD;
  float* router = w + o; o += (long)B_ * L_ * S_;
  float* simb   = w + o; o += (long)B_ * L_ * S_;
  float* sattn  = w + o; o += (long)B_ * L_ * S_;
  float* ent    = w + o; o += B_ * L_;
  float* enorm  = w + o; o += B_ * L_;
  float* cmean  = w + o; o += B_ * L_;
  float* cmax   = w + o; o += B_ * L_;
  float* dthr   = w + o; o += B_ * L_;
  float* util   = w + o; o += B_ * L_;
  float* encm   = w + o; o += B_ * D_;
  float* budg   = w + o; o += 32;
  float* wmask  = w + o; o += B_ * L_;
  float* impv   = w + o; o += B_ * M_;
  float* obsv   = w + o; o += B_ * M_;

  const float rsD  = 0.044194173824159216f;  // 1/sqrt(512)
  const float rsDh = 0.125f;                 // 1/sqrt(64)
  const int   TOK  = B_ * L_;                // 1024

  // ---- embedding ----
  embed_kernel<<<TOK, 256, 0, stream>>>(x, embW, pos, xemb, h);

  // ---- encoder layers ----
  for (int i = 0; i < NL_; ++i) {
    const float* Wq = qkvw + (long)i * 3 * D_ * D_;
    const float* bq = qkvb + (long)i * 3 * D_;
    // QKV: [1024,512] x [512,1536]
    launch_gemm(stream, h, D_, 0, 0, Wq, D_, 0, 0, bq,
                qkv, 3 * D_, 0, 0, TOK, 3 * D_, D_, 1, 1, 1, 0, 0, 1.f);
    // scores = Q K^T / sqrt(dh), z = b*H + head
    launch_gemm(stream,
                qkv,        3 * D_, (long)L_ * 3 * D_, 64,
                qkv + D_,   3 * D_, (long)L_ * 3 * D_, 64,
                nullptr,
                scores, L_, (long)H_ * L_ * L_, (long)L_ * L_,
                L_, L_, 64, B_ * H_, H_, 1, 0, 0, rsDh);
    softmax_causal_kernel<<<B_ * H_ * L_, 256, 0, stream>>>(scores);
    // O = P V
    launch_gemm(stream,
                scores, L_, (long)H_ * L_ * L_, (long)L_ * L_,
                qkv + 2 * D_, 3 * D_, (long)L_ * 3 * D_, 64,
                nullptr,
                attno, D_, (long)L_ * D_, 64,
                L_, 64, L_, B_ * H_, H_, 0, 0, 0, 1.f);
    // out proj
    launch_gemm(stream, attno, D_, 0, 0, outw + (long)i * D_ * D_, D_, 0, 0,
                outb + (long)i * D_, tmp, D_, 0, 0, TOK, D_, D_, 1, 1, 1, 0, 0, 1.f);
    ln_residual_kernel<<<TOK, 256, 0, stream>>>(h, tmp, ln1w + i * D_, ln1b + i * D_, h);
    // FFN
    launch_gemm(stream, h, D_, 0, 0, ff1w + (long)i * 4 * D_ * D_, D_, 0, 0,
                ff1b + (long)i * 4 * D_, ffh, 4 * D_, 0, 0, TOK, 4 * D_, D_, 1, 1, 1, 1, 0, 1.f);
    launch_gemm(stream, ffh, 4 * D_, 0, 0, ff2w + (long)i * D_ * 4 * D_, 4 * D_, 0, 0,
                ff2b + (long)i * D_, tmp, D_, 0, 0, TOK, D_, 4 * D_, 1, 1, 1, 0, 0, 1.f);
    ln_residual_kernel<<<TOK, 256, 0, stream>>>(h, tmp, ln2w + i * D_, ln2b + i * D_, h);
  }
  // h == encoded

  // ---- base logits (into d_out) + entropy ----
  launch_gemm(stream, h, D_, 0, 0, embW, D_, 0, 0, nullptr,
              out, V_, 0, 0, TOK, V_, D_, 1, 1, 1, 0, 0, 1.f);
  entropy_kernel<<<TOK, 256, 0, stream>>>(out, ent);
  entnorm_kernel<<<B_, 256, 0, stream>>>(ent, enorm);
  scan_kernel<<<B_, 32, 0, stream>>>(enorm, cmean, cmax);

  // ---- gates / projections ----
  small_mlp_kernel<<<TOK, 128, 0, stream>>>(h, D_, nullptr, nullptr,
      ug1w, ug1b, D_, 64, ug2w, ug2b, D_, unc, D_, 1, TOK);
  launch_gemm(stream, h, D_, 0, 0, qpw, D_, 0, 0, qpb, quer, D_, 0, 0,
              TOK, D_, D_, 1, 1, 1, 0, 0, 1.f);
  qscale_kernel<<<(int)((BLD + 255) / 256), 256, 0, stream>>>(quer, unc, (int)BLD);
  small_mlp_kernel<<<TOK, 128, 0, stream>>>(h, D_, cmean, cmax,
      thr1w, thr1b, D_, 64, thr2w, thr2b, 1, dthr, 1, 1, TOK);
  encmean_kernel<<<B_, 256, 0, stream>>>(h, encm);
  small_mlp_kernel<<<B_, 128, 0, stream>>>(encm, D_, nullptr, nullptr,
      bud1w, bud1b, D_, 32, bud2w, bud2b, 1, budg, 1, 1, B_);
  launch_gemm(stream, h, D_, 0, 0, kpw, D_, 0, 0, kpb, candk, D_, 0, 0,
              TOK, D_, D_, 1, 1, 1, 0, 0, 1.f);
  launch_gemm(stream, xemb, D_, 0, 0, vpw, D_, 0, 0, vpb, candv, D_, 0, 0,
              TOK, D_, D_, 1, 1, 1, 0, 0, 1.f);

  // ---- write gating stats (entropy_reg, l2_imp) ----
  zero2_kernel<<<1, 32, 0, stream>>>(outTail);
  mask_stats_kernel<<<TOK, 256, 0, stream>>>(enorm, dthr, candk, candv, wmask, outTail);

  // ---- retrieval attention ----
  // scores[:, :M] = quer . memk^T / sqrt(D)
  launch_gemm(stream,
              quer, D_, (long)L_ * D_, 0,
              memk, D_, (long)M_ * D_, 0, nullptr,
              scores, M_ + L_, (long)L_ * (M_ + L_), 0,
              L_, M_, D_, B_, 1, 1, 0, 0, rsD);
  // scores[:, M:] = quer . candk^T / sqrt(D)
  launch_gemm(stream,
              quer, D_, (long)L_ * D_, 0,
              candk, D_, (long)L_ * D_, 0, nullptr,
              scores + M_, M_ + L_, (long)L_ * (M_ + L_), 0,
              L_, L_, D_, B_, 1, 1, 0, 0, rsD);
  retr_softmax_kernel<<<TOK, 256, 0, stream>>>(scores, mage, wmask);
  // retrieved = attn[:, :M] . memv  +  attn[:, M:] . candv
  launch_gemm(stream,
              scores, M_ + L_, (long)L_ * (M_ + L_), 0,
              memv, D_, (long)M_ * D_, 0, nullptr,
              retr, D_, (long)L_ * D_, 0,
              L_, D_, M_, B_, 1, 0, 0, 0, 1.f);
  launch_gemm(stream,
              scores + M_, M_ + L_, (long)L_ * (M_ + L_), 0,
              candv, D_, (long)L_ * D_, 0, nullptr,
              retr, D_, (long)L_ * D_, 0,
              L_, D_, L_, B_, 1, 0, 0, 1, 1.f);
  launch_gemm(stream, retr, D_, 0, 0, fw, D_, 0, 0, fb, fuseo, D_, 0, 0,
              TOK, D_, D_, 1, 1, 1, 0, 0, 1.f);

  // ---- schema routing ----
  small_mlp_kernel<<<TOK, 128, 0, stream>>>(h, D_, nullptr, nullptr,
      rt1w, rt1b, D_, 64, rt2w, rt2b, S_, router, S_, 0, TOK);
  launch_gemm(stream, quer, D_, 0, 0, skeys, D_, 0, 0, nullptr,
              simb, S_, 0, 0, TOK, S_, D_, 1, 1, 1, 0, 0, rsD);
  schema_softmax_kernel<<<TOK, 64, 0, stream>>>(router, simb, sattn);
  launch_gemm(stream, sattn, S_, 0, 0, svals, D_, 0, 0, nullptr,
              scho, D_, 0, 0, TOK, D_, S_, 1, 1, 0, 0, 0, 1.f);
  small_mlp_kernel<<<TOK, 128, 0, stream>>>(h, D_, nullptr, nullptr,
      su1w, su1b, D_, 32, su2w, su2b, 1, util, 1, 0, TOK);
  schema_stats_kernel<<<S_, 256, 0, stream>>>(sattn, util, usum0, ucnt0, outTail);

  // ---- memory slot keep score ----
  small_mlp_kernel<<<B_ * M_, 128, 0, stream>>>(memv, D_, nullptr, nullptr,
      imp1w, imp1b, D_, 64, imp2w, imp2b, 1, impv, 1, 0, B_ * M_);
  small_mlp_kernel<<<B_ * M_, 128, 0, stream>>>(memv, D_, nullptr, nullptr,
      obs1w, obs1b, D_, 64, obs2w, obs2b, 1, obsv, 1, 0, B_ * M_);
  keep_kernel<<<(B_ * M_ + 255) / 256, 256, 0, stream>>>(impv, obsv, mage,
      outTail + 2 + 2 * S_, B_ * M_);

  // ---- final logits: d_out += (fused*budget + schema_out) @ embW^T ----
  delta_kernel<<<(int)((BLD + 255) / 256), 256, 0, stream>>>(fuseo, scho, budg, (int)BLD);
  launch_gemm(stream, fuseo, D_, 0, 0, embW, D_, 0, 0, nullptr,
              out, V_, 0, 0, TOK, V_, D_, 1, 1, 1, 0, 1, 1.f);
}
